// FullStepResonanceLayer_39058432589865
// MI455X (gfx1250) — matrix operand order repaired and checked by reading, not compiled, not run on hardware
//
#include <hip/hip_runtime.h>

// ---------------------------------------------------------------------------
// FullStepResonanceLayer for MI455X (gfx1250, wave32, WMMA).
// Kernel 1: fused time-embedding + ego*nei product + 3-layer MLP using
//           v_wmma_f32_16x16x32_f16, f16 weights/activations staged in LDS,
//           coalesced f32 tile writeback of f_re.
// Kernel 2: angle-partition statistics + W_ce projection -> re_matrix.
// ---------------------------------------------------------------------------

typedef __attribute__((ext_vector_type(16))) _Float16 v16h;
typedef __attribute__((ext_vector_type(8)))  _Float16 v8h;
typedef __attribute__((ext_vector_type(8)))  float    v8f;

#define Bsz 64
#define Nn  512
#define Tt  8
#define Dh  64          // D/2
#define Pp  8
#define TWO_PI_F 6.283185307179586f

#define LDK 136         // padded LDS row stride (halves): 272B = 68 dwords -> bank de-phase
#define ROWS_PER_BLOCK 128
#define TOTAL_ROWS (Bsz*Nn*Tt)          // 262144
#define RE_MATRIX_ELEMS (Bsz*Tt*Pp*128) // 524288

// LDS layout (in halves), floats appended after
#define W1_OFF 0
#define W2_OFF (128*LDK)
#define W3_OFF (2*128*LDK)
#define A0_OFF (2*128*LDK + 64*LDK)
#define A1_OFF (A0_OFF + 128*LDK)       // also reused as f32 staging for layer 3
#define HALF_TOTAL (A1_OFF + 128*LDK)
#define F_WTE 0
#define F_BTE 256
#define F_B1  384
#define F_B2  512
#define F_B3  640
#define FLOAT_TOTAL 704
#define SMEM_BYTES (HALF_TOTAL*2 + FLOAT_TOTAL*4)   // 159488 B < 320KB WGP LDS

// One wave computes a 16-row x outCols tile: out = relu(in[16x128] @ W + b).
// inAct: LDS f16 stride LDK. Wt: LDS f16 TRANSPOSED [outCol][k] stride LDK.
// OUT_F16: write f16 to outH (LDS, stride LDK); else f32 to outF (LDS, ld=64).
template <bool OUT_F16>
__device__ __forceinline__ void mlp_layer(
    const _Float16* inAct, const _Float16* Wt, const float* bias,
    _Float16* outH, float* outF, int outCols, int waveRow, int lane)
{
  const int l16  = lane & 15;
  const int half = lane >> 4;
  // A fragment (ISA 7.12.2, 16-bit A 16x32): lanes 0-15 -> M=l16, K {kb+0..7, kb+16..23}
  //                                          lanes16-31 -> M=l16, K {kb+8..15, kb+24..31}
  const _Float16* arow = inAct + (waveRow + l16) * LDK + half * 8;
  v16h Afrag[4];
#pragma unroll
  for (int kt = 0; kt < 4; ++kt) {
    v8h alo = *(const v8h*)(arow + kt * 32);
    v8h ahi = *(const v8h*)(arow + kt * 32 + 16);
    Afrag[kt] = __builtin_shufflevector(alo, ahi,
                  0,1,2,3,4,5,6,7,8,9,10,11,12,13,14,15);
  }
#pragma unroll 2
  for (int cb = 0; cb < outCols; cb += 16) {
    float bv = bias[cb + l16];                 // D layout: N = cb + l16 in every VGPR
    v8f acc = {bv, bv, bv, bv, bv, bv, bv, bv};
    // B fragment: lanes 0-15 K {kb+0..7, kb+16..23}, lanes 16-31 K {kb+8..15, kb+24..31}
    const _Float16* bcol = Wt + (cb + l16) * LDK + half * 8;
#pragma unroll
    for (int kt = 0; kt < 4; ++kt) {
      v8h blo = *(const v8h*)(bcol + kt * 32);
      v8h bhi = *(const v8h*)(bcol + kt * 32 + 16);
      v16h Bf = __builtin_shufflevector(blo, bhi,
                  0,1,2,3,4,5,6,7,8,9,10,11,12,13,14,15);
      acc = __builtin_amdgcn_wmma_f32_16x16x32_f16(
                false, Afrag[kt], false, Bf, (short)0, acc, false, false);
    }
#pragma unroll
    for (int r = 0; r < 8; ++r) {
      float v = fmaxf(acc[r], 0.0f);
      int rowr = waveRow + half * 8 + r;       // D layout: M = r (lanes<16) / 8+r
      if constexpr (OUT_F16) outH[rowr * LDK + cb + l16] = (_Float16)v;
      else                   outF[rowr * 64 + cb + l16] = v;
    }
  }
}

__global__ void __launch_bounds__(256, 1) resonance_mlp_kernel(
    const float* __restrict__ xe,  const float* __restrict__ xn,
    const float* __restrict__ Wte, const float* __restrict__ bte,
    const float* __restrict__ W1,  const float* __restrict__ b1,
    const float* __restrict__ W2,  const float* __restrict__ b2,
    const float* __restrict__ W3,  const float* __restrict__ b3,
    float* __restrict__ f_re)
{
  extern __shared__ char smem_raw[];
  _Float16* sH = (_Float16*)smem_raw;
  float*    sF = (float*)(smem_raw + HALF_TOTAL * 2);

  const int tid = threadIdx.x;

  // ---- stage weights: fp32 global -> f16 LDS, transposed [col][k] ----
  for (int idx = tid; idx < 128 * 128; idx += 256) {
    int k = idx >> 7, j = idx & 127;
    sH[W1_OFF + j * LDK + k] = (_Float16)W1[idx];
    sH[W2_OFF + j * LDK + k] = (_Float16)W2[idx];
  }
  for (int idx = tid; idx < 128 * 64; idx += 256) {
    int k = idx >> 6, j = idx & 63;
    sH[W3_OFF + j * LDK + k] = (_Float16)W3[idx];
  }
  sF[F_WTE + tid] = Wte[tid];                  // 256 elems, 256 threads
  if (tid < 128) { sF[F_BTE + tid] = bte[tid]; sF[F_B1 + tid] = b1[tid]; sF[F_B2 + tid] = b2[tid]; }
  if (tid < 64)  sF[F_B3 + tid] = b3[tid];
  __syncthreads();

  const int lane = tid & 31;
  const int wid  = tid >> 5;
  const int l16  = lane & 15;
  const int half = lane >> 4;
  const int waveRow  = wid * 16;
  const int rowLocal = waveRow + l16;
  const int blockRowBase = blockIdx.x * ROWS_PER_BLOCK;
  const int gRow = blockRowBase + rowLocal;

  // ---- phase 0: f = relu(ego_pure@Wte+bte) * relu(nei_pure@Wte+bte) -> A0 (f16) ----
  {
    int b   = gRow >> 12;         // / (N*T) = 4096
    int rem = gRow & 4095;
    int n   = rem >> 3;
    int t   = rem & 7;
    const float* e = xe + (size_t)(b * Tt) * 2;
    float ex = e[t * 2 + 0] - e[(Tt - 1) * 2 + 0];
    float ey = e[t * 2 + 1] - e[(Tt - 1) * 2 + 1];
    const float* xb = xn + (size_t)(b * Nn + n) * Tt * 2;
    float nx0 = xb[t * 2 + 0] - xb[(Tt - 1) * 2 + 0];
    float ny0 = xb[t * 2 + 1] - xb[(Tt - 1) * 2 + 1];
    int cbase = half * 64;        // lanes 0-15 cols 0..63, lanes 16-31 cols 64..127
    _Float16* dst = sH + A0_OFF + rowLocal * LDK + cbase;
    for (int c8 = 0; c8 < 64; c8 += 8) {
      v8h pk;
#pragma unroll
      for (int u = 0; u < 8; ++u) {
        int k = cbase + c8 + u;
        float w0 = sF[F_WTE + k], w1 = sF[F_WTE + 128 + k], bb = sF[F_BTE + k];
        float fe = fmaxf(ex * w0 + ey * w1 + bb, 0.0f);
        float fn = fmaxf(nx0 * w0 + ny0 * w1 + bb, 0.0f);
        pk[u] = (_Float16)(fe * fn);
      }
      *(v8h*)(dst + c8) = pk;     // ds_store_b128
    }
  }

  // ---- 3 WMMA layers; per-wave tiles, same-wave LDS ordering handles hazards ----
  float* sOut = (float*)(sH + A1_OFF);   // layer-3 f32 staging (128x64 = 32KB)
  mlp_layer<true >(sH + A0_OFF, sH + W1_OFF, sF + F_B1, sH + A1_OFF, nullptr, 128, waveRow, lane);
  mlp_layer<true >(sH + A1_OFF, sH + W2_OFF, sF + F_B2, sH + A0_OFF, nullptr, 128, waveRow, lane);
  __syncthreads();                       // A1 act reads done; safe to reuse as f32 staging
  mlp_layer<false>(sH + A0_OFF, sH + W3_OFF, sF + F_B3, nullptr, sOut, 64, waveRow, lane);
  __syncthreads();

  // ---- coalesced writeback of the 128x64 f32 tile (dominant HBM traffic) ----
  {
    const float4* src = (const float4*)sOut;
    float4* dstg = (float4*)(f_re + (size_t)blockRowBase * 64);
    for (int idx = tid; idx < ROWS_PER_BLOCK * 64 / 4; idx += 256)
      dstg[idx] = src[idx];              // global_store_b128, fully coalesced
  }
}

// ---- partition statistics: one block per (b,t) ----
__global__ void __launch_bounds__(256, 1) resonance_stats_kernel(
    const float* __restrict__ xe, const float* __restrict__ xn,
    const float* __restrict__ f_re,
    const float* __restrict__ Wce, const float* __restrict__ bce,
    float* __restrict__ re_matrix)
{
  __shared__ float cnt[Pp], sd[Pp], sa[Pp], sc[Pp * Dh];
  const int tid = threadIdx.x;
  const int b = blockIdx.x >> 3;
  const int t = blockIdx.x & 7;

  if (tid < Pp) { cnt[tid] = 0.f; sd[tid] = 0.f; sa[tid] = 0.f; }
  for (int i = tid; i < Pp * Dh; i += 256) sc[i] = 0.f;
  __syncthreads();

  const float elx = xe[(size_t)(b * Tt + Tt - 1) * 2 + 0];
  const float ely = xe[(size_t)(b * Tt + Tt - 1) * 2 + 1];
  const float etx = xe[(size_t)(b * Tt + t) * 2 + 0];
  const float ety = xe[(size_t)(b * Tt + t) * 2 + 1];

  for (int n = tid; n < Nn; n += 256) {
    const size_t xi = ((size_t)(b * Nn + n) * Tt + t) * 2;
    float px = xn[xi + 0] + elx - etx;
    float py = xn[xi + 1] + ely - ety;
    float dist = sqrtf(px * px + py * py);
    float ang = atan2f(px, py);               // reference: arctan2(p[...,0], p[...,1])
    if (ang < 0.f) ang += TWO_PI_F;
    int p = (int)(ang * ((float)Pp / TWO_PI_F));
    bool fm = (fabsf(px + py) > 1e-5f) && (dist > 0.005f);
    if (fm && p >= 0 && p < Pp) {
      atomicAdd(&cnt[p], 1.0f);
      atomicAdd(&sd[p], dist);
      atomicAdd(&sa[p], ang);
      const float* fr = f_re + ((size_t)(b * Nn + n) * Tt + t) * Dh;
#pragma unroll 8
      for (int c = 0; c < Dh; ++c) atomicAdd(&sc[p * Dh + c], fr[c]);
    }
  }
  __syncthreads();

  for (int idx = tid; idx < Pp * 128; idx += 256) {
    int p = idx >> 7, c = idx & 127;
    float nrm = cnt[p] + 1e-4f;
    float val;
    if (c < Dh) {
      val = sc[p * Dh + c] / nrm;
    } else {
      float dm = sd[p] / nrm, am = sa[p] / nrm;
      int j = c - Dh;
      val = fmaxf(dm * Wce[j] + am * Wce[Dh + j] + bce[j], 0.0f);
    }
    re_matrix[(((size_t)(b * Tt + t)) * Pp + p) * 128 + c] = val;
  }
}

extern "C" void kernel_launch(void* const* d_in, const int* in_sizes, int n_in,
                              void* d_out, int out_size, void* d_ws, size_t ws_size,
                              hipStream_t stream) {
  (void)in_sizes; (void)n_in; (void)out_size; (void)d_ws; (void)ws_size;
  const float* xe  = (const float*)d_in[0];
  const float* xn  = (const float*)d_in[1];
  const float* Wte = (const float*)d_in[2];
  const float* bte = (const float*)d_in[3];
  const float* W1  = (const float*)d_in[4];
  const float* b1  = (const float*)d_in[5];
  const float* W2  = (const float*)d_in[6];
  const float* b2  = (const float*)d_in[7];
  const float* W3  = (const float*)d_in[8];
  const float* b3  = (const float*)d_in[9];
  const float* Wce = (const float*)d_in[10];
  const float* bce = (const float*)d_in[11];

  float* out       = (float*)d_out;
  float* re_matrix = out;                     // [B,T,P,128]
  float* f_re      = out + RE_MATRIX_ELEMS;   // [B,N,T,64]

  resonance_mlp_kernel<<<TOTAL_ROWS / ROWS_PER_BLOCK, 256, SMEM_BYTES, stream>>>(
      xe, xn, Wte, bte, W1, b1, W2, b2, W3, b3, f_re);
  resonance_stats_kernel<<<Bsz * Tt, 256, 0, stream>>>(
      xe, xn, f_re, Wce, bce, re_matrix);
}
